// BERT_80805514707426
// MI455X (gfx1250) — compile-verified, matrix operand order checked by reading, not run on hardware
//
#include <hip/hip_runtime.h>
#include <hip/hip_bf16.h>
#include <math.h>

// ---------------------------------------------------------------------------
// BERT encoder forward for MI455X (gfx1250, wave32, WMMA f16 16x16x32).
// Reference dims: B=8 S=512 D=768 H=12 L=6 FF=3072 V=30522 C=2 HD=64.
// d_out = [ out(B*C) | attns(L*B*H*S*S) ]  (fp32, concatenated flat)
// ---------------------------------------------------------------------------

#define BB 8
#define SS 512
#define DD 768
#define HH 12
#define LL 6
#define FFD 3072
#define CC 2
#define HD 64
#define NTOK (BB * SS)          // 4096 rows
#define LN_EPS 1e-5f
#define ATTN_SCALE 0.125f       // 1/sqrt(64)

// WMMA tiling: block = 256 threads = 8 waves; each wave owns a 32x32 patch.
#define BM 128
#define BN 64
#define BK 64
#define LDA 72                  // padded As row stride (halves), 16B-aligned rows
#define LDB 72                  // padded Bs_t row stride (halves)

typedef __attribute__((ext_vector_type(16))) _Float16 v16h;
typedef __attribute__((ext_vector_type(8)))  _Float16 v8h;
typedef __attribute__((ext_vector_type(4)))  _Float16 v4h;
typedef __attribute__((ext_vector_type(8)))  float    v8f;
typedef __attribute__((ext_vector_type(4)))  float    v4f;

__device__ __forceinline__ v16h cat8(v8h lo, v8h hi) {
  return __builtin_shufflevector(lo, hi, 0, 1, 2, 3, 4, 5, 6, 7,
                                         8, 9, 10, 11, 12, 13, 14, 15);
}

// Two K=32 WMMA sub-steps (BK=64) for a wave's 32x32 patch (2x2 16x16 tiles).
// As: BM x BK (row-major, stride LDA). Bs: BN x BK (transposed, stride LDB).
// Per-lane fragments are two contiguous 8-half runs -> ds_load_b128 pairs.
__device__ __forceinline__ void mma_bk64(const _Float16* __restrict__ As,
                                         const _Float16* __restrict__ Bs,
                                         int lane, int wm, int wn,
                                         v8f (&acc)[2][2]) {
  const int l16 = lane & 15;
  const int kb  = (lane < 16) ? 0 : 8;
#pragma unroll
  for (int ks = 0; ks < BK; ks += 32) {
    const _Float16* a0p = As + (wm * 32 +      l16) * LDA + ks + kb;
    const _Float16* a1p = As + (wm * 32 + 16 + l16) * LDA + ks + kb;
    const _Float16* b0p = Bs + (wn * 32 +      l16) * LDB + ks + kb;
    const _Float16* b1p = Bs + (wn * 32 + 16 + l16) * LDB + ks + kb;
    const v16h a0 = cat8(*(const v8h*)a0p, *(const v8h*)(a0p + 16));
    const v16h a1 = cat8(*(const v8h*)a1p, *(const v8h*)(a1p + 16));
    const v16h b0 = cat8(*(const v8h*)b0p, *(const v8h*)(b0p + 16));
    const v16h b1 = cat8(*(const v8h*)b1p, *(const v8h*)(b1p + 16));
    acc[0][0] = __builtin_amdgcn_wmma_f32_16x16x32_f16(false, a0, false, b0, (short)0, acc[0][0], false, false);
    acc[0][1] = __builtin_amdgcn_wmma_f32_16x16x32_f16(false, a0, false, b1, (short)0, acc[0][1], false, false);
    acc[1][0] = __builtin_amdgcn_wmma_f32_16x16x32_f16(false, a1, false, b0, (short)0, acc[1][0], false, false);
    acc[1][1] = __builtin_amdgcn_wmma_f32_16x16x32_f16(false, a1, false, b1, (short)0, acc[1][1], false, false);
  }
}

// ---- split staging: global -> regs, regs -> LDS (for double buffering) ----

// A tile: BM x BK fp32 (row-major, row stride lda): 8 float4 per thread.
__device__ __forceinline__ void load_A_regs(const float* __restrict__ src,
                                            int lda, int tid, v4f (&f)[8]) {
#pragma unroll
  for (int p = 0; p < 8; ++p) {
    const int idx4 = tid + p * 256;
    const int r = idx4 >> 4, c4 = (idx4 & 15) << 2;
    f[p] = *(const v4f*)(src + (size_t)r * lda + c4);
  }
}
__device__ __forceinline__ void store_A_lds(_Float16* __restrict__ As,
                                            int tid, const v4f (&f)[8]) {
#pragma unroll
  for (int p = 0; p < 8; ++p) {
    const int idx4 = tid + p * 256;
    const int r = idx4 >> 4, c4 = (idx4 & 15) << 2;
    *(v4h*)(As + r * LDA + c4) = __builtin_convertvector(f[p], v4h);
  }
}

// B tile from row-major W[K,N] (row stride ldw): 4 float4 per thread,
// stored transposed into Bs_t[n][k] (4x b16 scatter stores).
__device__ __forceinline__ void load_B_regs(const float* __restrict__ src,
                                            int ldw, int tid, v4f (&f)[4]) {
#pragma unroll
  for (int p = 0; p < 4; ++p) {
    const int idx4 = tid + p * 256;
    const int r = idx4 >> 4, c4 = (idx4 & 15) << 2;   // r = k, c4 = n
    f[p] = *(const v4f*)(src + (size_t)r * ldw + c4);
  }
}
__device__ __forceinline__ void store_B_lds_T(_Float16* __restrict__ Bs,
                                              int tid, const v4f (&f)[4]) {
#pragma unroll
  for (int p = 0; p < 4; ++p) {
    const int idx4 = tid + p * 256;
    const int r = idx4 >> 4, c4 = (idx4 & 15) << 2;
#pragma unroll
    for (int i = 0; i < 4; ++i) Bs[(c4 + i) * LDB + r] = (_Float16)f[p][i];
  }
}

// K-matrix tile where source is [n][k] (row stride ldk): contiguous both ways.
__device__ __forceinline__ void stage_B_kmajor(const float* __restrict__ src,
                                               int ldk,
                                               _Float16* __restrict__ Bs,
                                               int tid) {
#pragma unroll
  for (int p = 0; p < 4; ++p) {
    const int idx4 = tid + p * 256;
    const int n = idx4 >> 4, k4 = (idx4 & 15) << 2;
    const v4f f = *(const v4f*)(src + (size_t)n * ldk + k4);
    *(v4h*)(Bs + n * LDB + k4) = __builtin_convertvector(f, v4h);
  }
}

// ---------------------------------------------------------------------------
// Generic GEMM: C[M,N] = epi(A[M,K] @ W[K,N] + bias[N]) (fp32 in memory,
// f16 WMMA), double-buffered LDS pipeline (1 barrier / K-slab).
// EPI: 0=bias, 1=bias+exact GELU, 2=bias+residual add.
// M % 128 == 0, N % 64 == 0, K % 64 == 0 (true for all call sites).
// ---------------------------------------------------------------------------
template <int EPI>
__global__ __launch_bounds__(256) void gemm_f32_wmma(
    const float* __restrict__ A, const float* __restrict__ W,
    const float* __restrict__ bias, float* __restrict__ C,
    const float* __restrict__ R, int M, int N, int K) {
  __shared__ _Float16 As[2][BM * LDA];
  __shared__ _Float16 Bs[2][BN * LDB];
  const int m0 = blockIdx.y * BM, n0 = blockIdx.x * BN;
  const int tid = threadIdx.x, lane = tid & 31;
  const int w = tid >> 5, wm = w >> 1, wn = w & 1;
  v8f acc[2][2] = {};
  v4f ar[8], br[4];

  load_A_regs(A + (size_t)m0 * K, K, tid, ar);
  load_B_regs(W + n0, N, tid, br);
  store_A_lds(As[0], tid, ar);
  store_B_lds_T(Bs[0], tid, br);
  __syncthreads();

  const int T = K / BK;
  for (int t = 0; t < T; ++t) {
    const int cur = t & 1;
    if (t + 1 < T) {                                  // prefetch next slab
      load_A_regs(A + (size_t)m0 * K + (t + 1) * BK, K, tid, ar);
      load_B_regs(W + (size_t)(t + 1) * BK * N + n0, N, tid, br);
    }
    mma_bk64(As[cur], Bs[cur], lane, wm, wn, acc);
    if (t + 1 < T) {
      store_A_lds(As[cur ^ 1], tid, ar);
      store_B_lds_T(Bs[cur ^ 1], tid, br);
      __syncthreads();
    }
  }

#pragma unroll
  for (int i = 0; i < 2; ++i)
#pragma unroll
    for (int j = 0; j < 2; ++j)
#pragma unroll
      for (int r = 0; r < 8; ++r) {
        const int mm = m0 + wm * 32 + i * 16 + r + ((lane >= 16) ? 8 : 0);
        const int nn = n0 + wn * 32 + j * 16 + (lane & 15);
        float val = acc[i][j][r] + bias[nn];
        if (EPI == 1) val = 0.5f * val * (1.0f + erff(val * 0.70710678118654752f));
        if (EPI == 2) val += R[(size_t)mm * N + nn];
        C[(size_t)mm * N + nn] = val;
      }
}

// ---------------------------------------------------------------------------
// Attention scores: per (b,h): out[q,k] = sum_d Q[q,d] * K[k,d]  (unscaled).
// Q,K stored [B,S,D] with head offset h*HD; out -> attn slice [B*H, S, S].
// K dimension = HD = 64 -> single BK slab, no K loop.
// ---------------------------------------------------------------------------
__global__ __launch_bounds__(256) void attn_scores_wmma(
    const float* __restrict__ q, const float* __restrict__ kmat,
    float* __restrict__ out) {
  __shared__ _Float16 As[BM * LDA];
  __shared__ _Float16 Bs[BN * LDB];
  const int bh = blockIdx.z, b = bh / HH, h = bh % HH;
  const int m0 = blockIdx.y * BM, n0 = blockIdx.x * BN;
  const int tid = threadIdx.x, lane = tid & 31;
  const int w = tid >> 5, wm = w >> 1, wn = w & 1;
  const float* qb = q    + (size_t)b * SS * DD + h * HD;
  const float* kb = kmat + (size_t)b * SS * DD + h * HD;
  float* ob = out + (size_t)bh * SS * SS;
  v8f acc[2][2] = {};
  v4f ar[8];
  load_A_regs(qb + (size_t)m0 * DD, DD, tid, ar);
  store_A_lds(As, tid, ar);
  stage_B_kmajor(kb + (size_t)n0 * DD, DD, Bs, tid);   // K[n][d]: k-major
  __syncthreads();
  mma_bk64(As, Bs, lane, wm, wn, acc);
#pragma unroll
  for (int i = 0; i < 2; ++i)
#pragma unroll
    for (int j = 0; j < 2; ++j)
#pragma unroll
      for (int r = 0; r < 8; ++r) {
        const int mm = m0 + wm * 32 + i * 16 + r + ((lane >= 16) ? 8 : 0);
        const int nn = n0 + wn * 32 + j * 16 + (lane & 15);
        ob[(size_t)mm * SS + nn] = acc[i][j][r];
      }
}

// ---------------------------------------------------------------------------
// attn @ V: per (b,h): o[q, n] = sum_k P[q,k] * V[k, n]. P fp32 (from d_out),
// V,[o] in [B,S,D] with head offset. N = HD = 64 (one BN tile).
// Double-buffered over K = S = 512 (8 slabs).
// ---------------------------------------------------------------------------
__global__ __launch_bounds__(256) void attn_v_wmma(
    const float* __restrict__ probs, const float* __restrict__ vmat,
    float* __restrict__ o) {
  __shared__ _Float16 As[2][BM * LDA];
  __shared__ _Float16 Bs[2][BN * LDB];
  const int bh = blockIdx.z, b = bh / HH, h = bh % HH;
  const int m0 = blockIdx.y * BM;
  const int tid = threadIdx.x, lane = tid & 31;
  const int w = tid >> 5, wm = w >> 1, wn = w & 1;
  const float* pb = probs + (size_t)bh * SS * SS;
  const float* vb = vmat + (size_t)b * SS * DD + h * HD;
  float* ob = o + (size_t)b * SS * DD + h * HD;
  v8f acc[2][2] = {};
  v4f ar[8], br[4];

  load_A_regs(pb + (size_t)m0 * SS, SS, tid, ar);
  load_B_regs(vb, DD, tid, br);
  store_A_lds(As[0], tid, ar);
  store_B_lds_T(Bs[0], tid, br);
  __syncthreads();

  const int T = SS / BK;                              // 8
  for (int t = 0; t < T; ++t) {
    const int cur = t & 1;
    if (t + 1 < T) {
      load_A_regs(pb + (size_t)m0 * SS + (t + 1) * BK, SS, tid, ar);
      load_B_regs(vb + (size_t)(t + 1) * BK * DD, DD, tid, br);
    }
    mma_bk64(As[cur], Bs[cur], lane, wm, wn, acc);
    if (t + 1 < T) {
      store_A_lds(As[cur ^ 1], tid, ar);
      store_B_lds_T(Bs[cur ^ 1], tid, br);
      __syncthreads();
    }
  }

#pragma unroll
  for (int i = 0; i < 2; ++i)
#pragma unroll
    for (int j = 0; j < 2; ++j)
#pragma unroll
      for (int r = 0; r < 8; ++r) {
        const int mm = m0 + wm * 32 + i * 16 + r + ((lane >= 16) ? 8 : 0);
        const int nn = wn * 32 + j * 16 + (lane & 15);
        ob[(size_t)mm * DD + nn] = acc[i][j][r];
      }
}

// ---------------------------------------------------------------------------
// Embedding + batch-indexed positional encoding (faithful to reference quirk).
// ---------------------------------------------------------------------------
__global__ __launch_bounds__(256) void embed_pe(
    const int* __restrict__ x, const float* __restrict__ emb,
    float* __restrict__ h) {
  const int idx = blockIdx.x * 256 + threadIdx.x;      // < B*S*D
  const int d = idx % DD, bs = idx / DD, s = bs % SS, b = bs / SS;
  const int tok = x[b * SS + s];
  const float d2 = (float)(d & ~1);
  const float div = __expf(-d2 * (9.210340371976184f / (float)DD)); // ln(1e4)
  const float ang = (float)b * div;
  const float pe = (d & 1) ? __cosf(ang) : __sinf(ang);
  h[idx] = emb[(size_t)tok * DD + d] + pe;
}

// ---------------------------------------------------------------------------
// LayerNorm over D=768 per row (one block per row, 256 threads).
// ---------------------------------------------------------------------------
__global__ __launch_bounds__(256) void layernorm_rows(
    const float* __restrict__ in, const float* __restrict__ g,
    const float* __restrict__ bta, float* __restrict__ out) {
  const int row = blockIdx.x, tid = threadIdx.x;
  const float* p = in + (size_t)row * DD;
  float* q = out + (size_t)row * DD;
  __shared__ float red[256];
  float s = 0.f;
  for (int i = tid; i < DD; i += 256) s += p[i];
  red[tid] = s; __syncthreads();
  for (int st = 128; st > 0; st >>= 1) { if (tid < st) red[tid] += red[tid + st]; __syncthreads(); }
  const float mean = red[0] / (float)DD; __syncthreads();
  float vs = 0.f;
  for (int i = tid; i < DD; i += 256) { const float dd = p[i] - mean; vs += dd * dd; }
  red[tid] = vs; __syncthreads();
  for (int st = 128; st > 0; st >>= 1) { if (tid < st) red[tid] += red[tid + st]; __syncthreads(); }
  const float inv = rsqrtf(red[0] / (float)DD + LN_EPS);
  for (int i = tid; i < DD; i += 256) q[i] = (p[i] - mean) * inv * g[i] + bta[i];
}

// ---------------------------------------------------------------------------
// In-place softmax over each S=512 row of the attn slice, with scale + mask.
// grid.x = B*H*S; block = 256, two elements per thread.
// ---------------------------------------------------------------------------
__global__ __launch_bounds__(256) void softmax_rows(
    float* __restrict__ scores, const int* __restrict__ mask) {
  const int row = blockIdx.x, tid = threadIdx.x;
  const int b = row / (HH * SS);
  float* p = scores + (size_t)row * SS;
  const int* mrow = mask + b * SS;
  __shared__ float red[256];
  float x0 = mrow[tid]       ? p[tid]       * ATTN_SCALE : -INFINITY;
  float x1 = mrow[tid + 256] ? p[tid + 256] * ATTN_SCALE : -INFINITY;
  red[tid] = fmaxf(x0, x1); __syncthreads();
  for (int st = 128; st > 0; st >>= 1) { if (tid < st) red[tid] = fmaxf(red[tid], red[tid + st]); __syncthreads(); }
  const float mx = red[0]; __syncthreads();
  const float e0 = __expf(x0 - mx), e1 = __expf(x1 - mx);
  red[tid] = e0 + e1; __syncthreads();
  for (int st = 128; st > 0; st >>= 1) { if (tid < st) red[tid] += red[tid + st]; __syncthreads(); }
  const float rs = 1.0f / red[0];
  p[tid] = e0 * rs; p[tid + 256] = e1 * rs;
}

// ---------------------------------------------------------------------------
// CLS head: LN(h[b,0,:]) @ Wc + bc  ->  out[b, 0:2]. One block per batch.
// ---------------------------------------------------------------------------
__global__ __launch_bounds__(256) void cls_head(
    const float* __restrict__ hmat, const float* __restrict__ g,
    const float* __restrict__ bta, const float* __restrict__ Wc,
    const float* __restrict__ bc, float* __restrict__ out) {
  const int b = blockIdx.x, tid = threadIdx.x;
  const float* p = hmat + (size_t)b * SS * DD;   // seq position 0
  __shared__ float red[256];
  float s = 0.f;
  for (int i = tid; i < DD; i += 256) s += p[i];
  red[tid] = s; __syncthreads();
  for (int st = 128; st > 0; st >>= 1) { if (tid < st) red[tid] += red[tid + st]; __syncthreads(); }
  const float mean = red[0] / (float)DD; __syncthreads();
  float vs = 0.f;
  for (int i = tid; i < DD; i += 256) { const float dd = p[i] - mean; vs += dd * dd; }
  red[tid] = vs; __syncthreads();
  for (int st = 128; st > 0; st >>= 1) { if (tid < st) red[tid] += red[tid + st]; __syncthreads(); }
  const float inv = rsqrtf(red[0] / (float)DD + LN_EPS); __syncthreads();
  float a0 = 0.f, a1 = 0.f;
  for (int i = tid; i < DD; i += 256) {
    const float nv = (p[i] - mean) * inv * g[i] + bta[i];
    a0 += nv * Wc[i * CC + 0];
    a1 += nv * Wc[i * CC + 1];
  }
  red[tid] = a0; __syncthreads();
  for (int st = 128; st > 0; st >>= 1) { if (tid < st) red[tid] += red[tid + st]; __syncthreads(); }
  const float r0 = red[0]; __syncthreads();
  red[tid] = a1; __syncthreads();
  for (int st = 128; st > 0; st >>= 1) { if (tid < st) red[tid] += red[tid + st]; __syncthreads(); }
  if (tid == 0) { out[b * CC + 0] = r0 + bc[0]; out[b * CC + 1] = red[0] + bc[1]; }
}

// ---------------------------------------------------------------------------
extern "C" void kernel_launch(void* const* d_in, const int* in_sizes, int n_in,
                              void* d_out, int out_size, void* d_ws, size_t ws_size,
                              hipStream_t stream) {
  (void)in_sizes; (void)n_in; (void)out_size; (void)ws_size;
  // Inputs (setup_inputs order)
  const int*   x     = (const int*)  d_in[0];
  const int*   mask  = (const int*)  d_in[1];
  const float* emb   = (const float*)d_in[2];
  const float* Wq    = (const float*)d_in[3];
  const float* bq    = (const float*)d_in[4];
  const float* Wk    = (const float*)d_in[5];
  const float* bk    = (const float*)d_in[6];
  const float* Wv    = (const float*)d_in[7];
  const float* bv    = (const float*)d_in[8];
  const float* Wo    = (const float*)d_in[9];
  const float* bo    = (const float*)d_in[10];
  const float* ln1_g = (const float*)d_in[11];
  const float* ln1_b = (const float*)d_in[12];
  const float* ln2_g = (const float*)d_in[13];
  const float* ln2_b = (const float*)d_in[14];
  const float* W1    = (const float*)d_in[15];
  const float* bf1   = (const float*)d_in[16];
  const float* W2    = (const float*)d_in[17];
  const float* bf2   = (const float*)d_in[18];
  const float* lnc_g = (const float*)d_in[19];
  const float* lnc_b = (const float*)d_in[20];
  const float* Wc    = (const float*)d_in[21];
  const float* bc    = (const float*)d_in[22];

  float* outp = (float*)d_out;                      // [0,16) = logits
  float* attn_base = outp + BB * CC;                // then [L,B,H,S,S]

  // Workspace layout (fp32): h, hn, q, k, v, o (B*S*D each) + ffh (B*S*FF)
  float* ws = (float*)d_ws;
  const size_t ND = (size_t)NTOK * DD;
  float* h   = ws;
  float* hn  = h  + ND;
  float* q   = hn + ND;
  float* kb_ = q  + ND;
  float* v   = kb_ + ND;
  float* o   = v  + ND;
  float* ffh = o  + ND;                             // NTOK*FF floats

  const dim3 blk(256);
  const dim3 g_gemm_d (DD  / BN, NTOK / BM);        // (12, 32)
  const dim3 g_gemm_ff(FFD / BN, NTOK / BM);        // (48, 32)
  const dim3 g_scores (SS / BN, SS / BM, BB * HH);  // (8, 4, 96)
  const dim3 g_av     (1,        SS / BM, BB * HH); // (1, 4, 96)

  embed_pe<<<(NTOK * DD) / 256, blk, 0, stream>>>(x, emb, h);

  for (int l = 0; l < LL; ++l) {
    float* attn_l = attn_base + (size_t)l * BB * HH * SS * SS;
    const size_t wD = (size_t)l * DD * DD, bD = (size_t)l * DD;
    const size_t wF1 = (size_t)l * DD * FFD, bF = (size_t)l * FFD;
    const size_t wF2 = (size_t)l * FFD * DD;

    layernorm_rows<<<NTOK, blk, 0, stream>>>(h, ln1_g + bD, ln1_b + bD, hn);
    gemm_f32_wmma<0><<<g_gemm_d, blk, 0, stream>>>(hn, Wq + wD, bq + bD, q,   nullptr, NTOK, DD, DD);
    gemm_f32_wmma<0><<<g_gemm_d, blk, 0, stream>>>(hn, Wk + wD, bk + bD, kb_, nullptr, NTOK, DD, DD);
    gemm_f32_wmma<0><<<g_gemm_d, blk, 0, stream>>>(hn, Wv + wD, bv + bD, v,   nullptr, NTOK, DD, DD);

    attn_scores_wmma<<<g_scores, blk, 0, stream>>>(q, kb_, attn_l);
    softmax_rows<<<BB * HH * SS, blk, 0, stream>>>(attn_l, mask);
    attn_v_wmma<<<g_av, blk, 0, stream>>>(attn_l, v, o);

    // h = h + o @ Wo + bo   (residual epilogue, in-place safe per tile)
    gemm_f32_wmma<2><<<g_gemm_d, blk, 0, stream>>>(o, Wo + wD, bo + bD, h, h, NTOK, DD, DD);

    layernorm_rows<<<NTOK, blk, 0, stream>>>(h, ln2_g + bD, ln2_b + bD, hn);
    gemm_f32_wmma<1><<<g_gemm_ff, blk, 0, stream>>>(hn, W1 + wF1, bf1 + bF, ffh, nullptr, NTOK, FFD, DD);
    gemm_f32_wmma<2><<<g_gemm_d,  blk, 0, stream>>>(ffh, W2 + wF2, bf2 + bD, h, h, NTOK, DD, FFD);
  }

  cls_head<<<BB, blk, 0, stream>>>(h, lnc_g, lnc_b, Wc, bc, outp);
}
